// DynamicEncoder_63342177682106
// MI455X (gfx1250) — matrix operand order repaired and checked by reading, not compiled, not run on hardware
//
#include <hip/hip_runtime.h>
#include <hip/hip_bf16.h>

typedef __attribute__((ext_vector_type(16))) _Float16 v16h;
typedef __attribute__((ext_vector_type(8)))  _Float16 v8h;
typedef __attribute__((ext_vector_type(8)))  float    v8f;
typedef __attribute__((ext_vector_type(4)))  float    v4f;

#define NBATCH   64
#define T_INX    2047     // raw x time length (a zero frame is appended -> 2048)
#define C0_IN    100      // layer-1 input channels
#define C0_PAD   128      // padded to 4 x K32 chunks
#define T1       1024     // layer-1 output times
#define C1       112      // layer-1 output channels (pool-fused)
#define C1_PAD   128
#define T2       512      // layer-2 output times
#define KTAPS    15
#define MT       7        // 112 / 16 output-channel tiles
#define SLOTS    80       // staged input time slots per block (need 2*31+14+1 = 77)
#define FRAG_N   (MT*KTAPS*4*512)   // 215040 f16 elements per weight-frag buffer

// ---- graph constants derived from SMPL_EDGES (hardcoded; they are not inputs) ----
// distance<=1 neighbor bitmask over 25 layer-1 entities (23 edges + 2 globals)
__device__ const unsigned nb_mask0[25] = {
    0x0180000Fu, 0x01800017u, 0x01800027u, 0x01800049u,
    0x00000092u, 0x00000124u, 0x00000248u, 0x00000490u,
    0x00003920u, 0x00000240u, 0x00000480u, 0x00007900u,
    0x0000B900u, 0x00013900u, 0x00004800u, 0x00029000u,
    0x00052000u, 0x000A8000u, 0x00150000u, 0x002A0000u,
    0x00540000u, 0x00280000u, 0x00500000u, 0x0180000Fu,
    0x0180000Fu
};
// distance<=1 neighbor bitmask over 14 layer-2 entities (13 pooled edges + 1 global)
__device__ const unsigned nb_mask1[14] = {
    0x2017u, 0x2003u, 0x201Du, 0x000Cu, 0x2035u, 0x04F0u, 0x04E0u,
    0x05E0u, 0x0380u, 0x0300u, 0x0CE0u, 0x1C00u, 0x1800u, 0x2017u
};
// P0 pools: 14 pools over 25 entities (8 channels each)
__device__ const int pool0_j[14][2] = {
    {0,3},{6,9},{1,4},{7,10},{2,0},{5,8},{11,14},
    {12,0},{15,17},{19,21},{13,0},{16,18},{20,22},{23,24}
};
__device__ const int pool0_n[14] = {2,2,2,2,1,2,2,1,2,2,1,2,2,2};
// P1 pools: 7 pools over 14 entities (16 channels each)
__device__ const int pool1_j[7][3] = {
    {0,1,0},{2,3,0},{4,5,0},{6,0,0},{7,8,9},{10,11,12},{13,0,0}
};
__device__ const int pool1_n[7] = {2,2,2,1,3,3,1};

// A-fragment K order: contiguous k = 16*(lane/16) + h (matches B fragment)
__device__ __forceinline__ int kmapA(int lane, int h) {
    return h + ((lane & 16) ? 16 : 0);
}

// ---------------- prep: build pool+mask-fused weights in WMMA A-fragment layout ----
__global__ void prep_w0_frag(const float* __restrict__ w0, const float* __restrict__ b0,
                             _Float16* __restrict__ frag, float* __restrict__ beff) {
    int idx = blockIdx.x * 256 + threadIdx.x;
    if (idx < C1) {                       // fused bias: b0eff = P0 @ b0
        int p = idx >> 3, r = idx & 7;
        float s = 0.f;
        for (int i = 0; i < pool0_n[p]; ++i) s += b0[pool0_j[p][i] * 8 + r];
        beff[idx] = s / (float)pool0_n[p];
    }
    if (idx >= FRAG_N) return;
    int h    = idx & 15;
    int lane = (idx >> 4) & 31;
    int rest = idx >> 9;                  // (mt*15 + k)*4 + c
    int c    = rest & 3;
    int kk   = rest >> 2;
    int k    = kk % KTAPS;
    int mt   = kk / KTAPS;
    int m    = mt * 16 + (lane & 15);     // pooled output channel 0..111
    int kidx = c * 32 + kmapA(lane, h);   // padded input channel 0..127
    float val = 0.f;
    if (kidx < C0_IN) {
        int p = m >> 3, r = m & 7;
        int nn = pool0_n[p];
        float s = 0.f;
        for (int i = 0; i < nn; ++i) {
            int j = pool0_j[p][i];
            if ((nb_mask0[j] >> (kidx >> 2)) & 1u)
                s += w0[(size_t)(j * 8 + r) * (C0_IN * KTAPS) + kidx * KTAPS + k];
        }
        val = s / (float)nn;
    }
    frag[idx] = (_Float16)val;
}

__global__ void prep_w1_frag(const float* __restrict__ w1, const float* __restrict__ b1,
                             _Float16* __restrict__ frag, float* __restrict__ beff) {
    int idx = blockIdx.x * 256 + threadIdx.x;
    if (idx < C1) {                       // fused bias: b1eff = P1 @ b1
        int q = idx >> 4, r = idx & 15;
        float s = 0.f;
        for (int i = 0; i < pool1_n[q]; ++i) s += b1[pool1_j[q][i] * 16 + r];
        beff[idx] = s / (float)pool1_n[q];
    }
    if (idx >= FRAG_N) return;
    int h    = idx & 15;
    int lane = (idx >> 4) & 31;
    int rest = idx >> 9;
    int c    = rest & 3;
    int kk   = rest >> 2;
    int k    = kk % KTAPS;
    int mt   = kk / KTAPS;
    int m    = mt * 16 + (lane & 15);
    int kidx = c * 32 + kmapA(lane, h);
    float val = 0.f;
    if (kidx < C1) {
        int q = m >> 4, r = m & 15;
        int nn = pool1_n[q];
        float s = 0.f;
        for (int i = 0; i < nn; ++i) {
            int j = pool1_j[q][i];
            if ((nb_mask1[j] >> (kidx >> 3)) & 1u)
                s += w1[(size_t)(j * 16 + r) * (C1 * KTAPS) + kidx * KTAPS + k];
        }
        val = s / (float)nn;
    }
    frag[idx] = (_Float16)val;
}

// ---------------- layer 1: implicit-GEMM conv (stride 2, pad 7) + fused pool/bias/relu
// block: 224 threads = 7 waves, one wave per 16-channel M tile over TWO 16-time N tiles;
// each A fragment load feeds two WMMAs. One block per (32-time chunk, batch).
__global__ __launch_bounds__(224) void layer1_conv(
    const float* __restrict__ x, const _Float16* __restrict__ wfrag,
    const float* __restrict__ beff, _Float16* __restrict__ act1)
{
    __shared__ _Float16 lds[SLOTS * C0_PAD];          // 20 KB
    const int t0  = blockIdx.x * 32;                  // output time chunk base
    const int b   = blockIdx.y;
    const int tid = threadIdx.x;

    // stage input times [2*t0-7, 2*t0+72], channels padded to 128
    for (int task = tid; task < SLOTS * (C0_PAD / 4); task += 224) {
        int s = task >> 5;                            // time slot 0..79
        int g = task & 31;                            // channel group of 4
        int it = 2 * t0 - 7 + s;                      // index into 2048-long padded seq
        v4f v = {};
        if (g < 25 && it >= 0 && it < T_INX)          // it==2047 is the appended zero row
            v = *(const v4f*)(x + ((size_t)b * T_INX + it) * C0_IN + g * 4);
        _Float16* d = lds + s * C0_PAD + g * 4;
        d[0] = (_Float16)v.x; d[1] = (_Float16)v.y;
        d[2] = (_Float16)v.z; d[3] = (_Float16)v.w;
    }
    __syncthreads();

    const int wave = tid >> 5;                        // M tile 0..6
    const int lane = tid & 31;
    const int n    = lane & 15;                       // output-time column
    const int khi  = (lane & 16) ? 16 : 0;            // B-fragment K offset
    const _Float16* wbase = wfrag + (size_t)wave * KTAPS * 4 * 512 + lane * 16;

    v8f acc0 = {}, acc1 = {};
    for (int k = 0; k < KTAPS; ++k) {
        const _Float16* brow = lds + (2 * n + k) * C0_PAD + khi;
#pragma unroll
        for (int c = 0; c < 4; ++c) {
            v16h a  = *(const v16h*)(wbase + (size_t)(k * 4 + c) * 512);
            v16h bA = *(const v16h*)(brow + c * 32);
            v16h bB = *(const v16h*)(brow + 32 * C0_PAD + c * 32);
            acc0 = __builtin_amdgcn_wmma_f32_16x16x32_f16(false, a, false, bA,
                                                          (short)0, acc0, false, false);
            acc1 = __builtin_amdgcn_wmma_f32_16x16x32_f16(false, a, false, bB,
                                                          (short)0, acc1, false, false);
        }
    }

    const int m0 = wave * 16 + ((lane & 16) ? 8 : 0); // D: m = m0 + v
    v8f bia = *(const v8f*)(beff + m0);
    v8h oh0, oh1;
#pragma unroll
    for (int v = 0; v < 8; ++v) {
        float f0 = acc0[v] + bia[v];
        float f1 = acc1[v] + bia[v];
        oh0[v] = (_Float16)(f0 > 0.f ? f0 : 0.f);
        oh1[v] = (_Float16)(f1 > 0.f ? f1 : 0.f);
    }
    // act1 layout: [b][time 1024][ch 112] f16 ; 8 contiguous channels per lane
    *(v8h*)(act1 + ((size_t)b * T1 + (t0 + n)) * C1 + m0)      = oh0;
    *(v8h*)(act1 + ((size_t)b * T1 + (t0 + 16 + n)) * C1 + m0) = oh1;
}

// ---------------- layer 2: same structure, f16 activations in, f32 out + relu
__global__ __launch_bounds__(224) void layer2_conv(
    const _Float16* __restrict__ act1, const _Float16* __restrict__ wfrag,
    const float* __restrict__ beff, float* __restrict__ out)
{
    __shared__ _Float16 lds[SLOTS * C1_PAD];          // 20 KB
    const int t0  = blockIdx.x * 32;
    const int b   = blockIdx.y;
    const int tid = threadIdx.x;

    for (int task = tid; task < SLOTS * (C1_PAD / 8); task += 224) {
        int s = task >> 4;                            // time slot 0..79
        int g = task & 15;                            // channel group of 8
        int it = 2 * t0 - 7 + s;
        v8h v = {};
        if (g < 14 && it >= 0 && it < T1)
            v = *(const v8h*)(act1 + ((size_t)b * T1 + it) * C1 + g * 8);
        *(v8h*)(lds + s * C1_PAD + g * 8) = v;
    }
    __syncthreads();

    const int wave = tid >> 5;
    const int lane = tid & 31;
    const int n    = lane & 15;
    const int khi  = (lane & 16) ? 16 : 0;
    const _Float16* wbase = wfrag + (size_t)wave * KTAPS * 4 * 512 + lane * 16;

    v8f acc0 = {}, acc1 = {};
    for (int k = 0; k < KTAPS; ++k) {
        const _Float16* brow = lds + (2 * n + k) * C1_PAD + khi;
#pragma unroll
        for (int c = 0; c < 4; ++c) {
            v16h a  = *(const v16h*)(wbase + (size_t)(k * 4 + c) * 512);
            v16h bA = *(const v16h*)(brow + c * 32);
            v16h bB = *(const v16h*)(brow + 32 * C1_PAD + c * 32);
            acc0 = __builtin_amdgcn_wmma_f32_16x16x32_f16(false, a, false, bA,
                                                          (short)0, acc0, false, false);
            acc1 = __builtin_amdgcn_wmma_f32_16x16x32_f16(false, a, false, bB,
                                                          (short)0, acc1, false, false);
        }
    }

    const int m0 = wave * 16 + ((lane & 16) ? 8 : 0);
    v8f bia = *(const v8f*)(beff + m0);
    const int t = t0 + n;
#pragma unroll
    for (int v = 0; v < 8; ++v) {
        float f0 = acc0[v] + bia[v];
        float f1 = acc1[v] + bia[v];
        f0 = f0 > 0.f ? f0 : 0.f;
        f1 = f1 > 0.f ? f1 : 0.f;
        out[((size_t)b * C1 + (m0 + v)) * T2 + t]      = f0;  // (b, p, t) layout
        out[((size_t)b * C1 + (m0 + v)) * T2 + t + 16] = f1;
    }
}

// ---------------- launch ----------------
extern "C" void kernel_launch(void* const* d_in, const int* in_sizes, int n_in,
                              void* d_out, int out_size, void* d_ws, size_t ws_size,
                              hipStream_t stream) {
    (void)in_sizes; (void)n_in; (void)out_size; (void)ws_size;
    const float* x  = (const float*)d_in[0];
    const float* w0 = (const float*)d_in[1];
    const float* b0 = (const float*)d_in[2];
    const float* w1 = (const float*)d_in[3];
    const float* b1 = (const float*)d_in[4];
    float* out = (float*)d_out;

    char* ws = (char*)d_ws;
    _Float16* w0f  = (_Float16*)(ws + 0);          // 430080 B
    _Float16* w1f  = (_Float16*)(ws + 430080);     // 430080 B
    float*    b0e  = (float*)   (ws + 860160);     // 448 B
    float*    b1e  = (float*)   (ws + 860672);     // 448 B
    _Float16* act1 = (_Float16*)(ws + 861184);     // 64*1024*112*2 = 14.7 MB

    prep_w0_frag<<<FRAG_N / 256, 256, 0, stream>>>(w0, b0, w0f, b0e);
    prep_w1_frag<<<FRAG_N / 256, 256, 0, stream>>>(w1, b1, w1f, b1e);
    layer1_conv<<<dim3(T1 / 32, NBATCH), 224, 0, stream>>>(x, w0f, b0e, act1);
    layer2_conv<<<dim3(T2 / 32, NBATCH), 224, 0, stream>>>(act1, w1f, b1e, out);
}
